// SelfAttention_41274635714819
// MI455X (gfx1250) — matrix operand order repaired and checked by reading, not compiled, not run on hardware
//
#include <hip/hip_runtime.h>
#include <hip/hip_bf16.h>

typedef _Float16 f16;
typedef __attribute__((ext_vector_type(16))) _Float16 v16h;
typedef __attribute__((ext_vector_type(8)))  _Float16 v8h;
typedef __attribute__((ext_vector_type(8)))  float    v8f;

#define DIM  1024
#define LSEQ 2048
#define NB   4
#define NH   16
#define HD   64

// ---- CDNA5 async global->LDS path (ASYNCcnt), via inline asm --------------
// global_load_async_to_lds_b128: VDST = 32-bit LDS byte offset, VADDR = 64-bit
// global address (GV mode, SADDR = off).  Tracked by ASYNCcnt (ISA 15.18.3).
__device__ inline unsigned lds_off(const f16* p) {
    return (unsigned)(size_t)(__attribute__((address_space(3))) const f16*)p;
}
__device__ inline void async_cp16(f16* ldst, const f16* gsrc) {
    unsigned l = lds_off(ldst);
    asm volatile("global_load_async_to_lds_b128 %0, %1, off"
                 :: "v"(l), "v"(gsrc) : "memory");
}
__device__ inline void async_wait8() {
    asm volatile("s_wait_asynccnt 8" ::: "memory");
}
__device__ inline void async_wait0() {
    asm volatile("s_wait_asynccnt 0" ::: "memory");
}

// ---- WMMA helpers (layouts per ISA 7.12.2, wave32) ------------------------
__device__ inline v8f wmma_f16(v16h a, v16h b, v8f c) {
    return __builtin_amdgcn_wmma_f32_16x16x32_f16(false, a, false, b,
                                                  (short)0, c, false, false);
}
// A 16x32: row = lane&15 ; k=hi*16+e
__device__ inline v16h load_a_frag(const f16* row_ptr, int lane, int koff) {
    const f16* p = row_ptr + koff + ((lane >> 4) & 1) * 16;
    v16h a;
#pragma unroll
    for (int i = 0; i < 16; ++i) a[i] = p[i];
    return a;
}
// B 32x16 from LDS [N][K] (K contiguous per column)
__device__ inline v16h load_b_frag(const f16* nk, int lane, int strideK, int n0, int koff) {
    const f16* p = nk + (size_t)(n0 + (lane & 15)) * strideK + koff + ((lane >> 4) & 1) * 16;
    v16h b;
#pragma unroll
    for (int i = 0; i < 16; ++i) b[i] = p[i];
    return b;
}

// ---------------------------------------------------------------------------
// Kernel 1: qkv = x @ W_qkv, fused RMSNorm + RoPE epilogue.
// Block tile 128(M) x 64(N), K-step 32; each wave owns 32 rows (8 WMMA/step).
// q,k written [B,H,L,D]; v written TRANSPOSED [B,H,D,L] for the PV B-tile.
// ---------------------------------------------------------------------------
__global__ __launch_bounds__(128) void qkv_kernel(
    const float* __restrict__ x, const float* __restrict__ Wqkv,
    const float* __restrict__ pe, const float* __restrict__ qs,
    const float* __restrict__ ks,
    f16* __restrict__ qb, f16* __restrict__ kb, f16* __restrict__ vb)
{
    __shared__ __align__(16) f16 la[128 * 32];         // [m][k]
    __shared__ __align__(16) f16 lb[64 * 32];          // [n][k] (transposed)
    __shared__ __align__(16) float ct[128 * 65];       // f32 tile for epilogue

    const int tid = threadIdx.x, lane = tid & 31, w = tid >> 5;
    const int m0 = blockIdx.x * 128;                   // flat row (b*L + l)
    const int n0 = blockIdx.y * 64;                    // col in [0,3072)

    v8f acc[2][4] = {};
    for (int k0 = 0; k0 < DIM; k0 += 32) {
        if (k0 + 32 < DIM)
            __builtin_prefetch(Wqkv + (size_t)(k0 + 32) * 3072 + n0, 0, 3);
        // stage A: 128x32 f32 -> f16
        for (int i = tid * 4; i < 128 * 32; i += 128 * 4) {
            int r = i >> 5, c = i & 31;
            float4 f = *(const float4*)(x + (size_t)(m0 + r) * DIM + k0 + c);
            la[r * 32 + c + 0] = (f16)f.x; la[r * 32 + c + 1] = (f16)f.y;
            la[r * 32 + c + 2] = (f16)f.z; la[r * 32 + c + 3] = (f16)f.w;
        }
        // stage B transposed: W[k][n] -> lb[n][k]
        for (int i = tid * 4; i < 32 * 64; i += 128 * 4) {
            int r = i >> 6, c = i & 63;
            float4 f = *(const float4*)(Wqkv + (size_t)(k0 + r) * 3072 + n0 + c);
            lb[(c + 0) * 32 + r] = (f16)f.x; lb[(c + 1) * 32 + r] = (f16)f.y;
            lb[(c + 2) * 32 + r] = (f16)f.z; lb[(c + 3) * 32 + r] = (f16)f.w;
        }
        __syncthreads();
        v16h a0 = load_a_frag(&la[(w * 32 +      (lane & 15)) * 32], lane, 0);
        v16h a1 = load_a_frag(&la[(w * 32 + 16 + (lane & 15)) * 32], lane, 0);
#pragma unroll
        for (int t = 0; t < 4; ++t) {
            v16h b = load_b_frag(lb, lane, 32, t * 16, 0);
            acc[0][t] = wmma_f16(a0, b, acc[0][t]);
            acc[1][t] = wmma_f16(a1, b, acc[1][t]);
        }
        __syncthreads();
    }
    const int hi8 = (lane >> 4) * 8;
#pragma unroll
    for (int u = 0; u < 2; ++u)
#pragma unroll
        for (int t = 0; t < 4; ++t)
#pragma unroll
            for (int r = 0; r < 8; ++r)
                ct[(w * 32 + u * 16 + r + hi8) * 65 + t * 16 + (lane & 15)] = acc[u][t][r];
    __syncthreads();

    {   // one row per thread: RMSNorm + RoPE + scatter
        const int row = m0 + tid;
        const int b = row / LSEQ, l = row % LSEQ;
        const int slot = n0 / DIM;             // 0=q 1=k 2=v
        const int h = (n0 % DIM) / HD;
        float tv[64];
#pragma unroll
        for (int d = 0; d < 64; ++d) tv[d] = ct[tid * 65 + d];

        if (slot == 2) {
            const size_t dstT = (((size_t)b * NH + h) * HD) * LSEQ + l;  // [B,H,D,L]
#pragma unroll
            for (int d = 0; d < 64; ++d) vb[dstT + (size_t)d * LSEQ] = (f16)tv[d];
        } else {
            const size_t dst = (((size_t)b * NH + h) * LSEQ + l) * HD;   // [B,H,L,D]
            float ss = 0.f;
#pragma unroll
            for (int d = 0; d < 64; ++d) ss += tv[d] * tv[d];
            const float rr = rsqrtf(ss * (1.0f / 64.0f) + 1e-6f);
            const float* sc = (slot == 0) ? qs : ks;
            const float* pel = pe + (size_t)l * 128;   // [32][2][2]
            const float mul = (slot == 0) ? 0.125f : 1.0f;  // fold 1/sqrt(D) into q
            f16* outp = (slot == 0) ? qb : kb;
#pragma unroll
            for (int j = 0; j < 32; ++j) {
                float x0 = tv[2 * j] * rr * sc[2 * j];
                float x1 = tv[2 * j + 1] * rr * sc[2 * j + 1];
                outp[dst + 2 * j]     = (f16)((pel[j * 4 + 0] * x0 + pel[j * 4 + 1] * x1) * mul);
                outp[dst + 2 * j + 1] = (f16)((pel[j * 4 + 2] * x0 + pel[j * 4 + 3] * x1) * mul);
            }
        }
    }
}

// ---------------------------------------------------------------------------
// Kernel 2: flash attention, one (b,h,64-query) tile per block.
// K/V tiles double-buffered via GLOBAL_LOAD_ASYNC_TO_LDS (ASYNCcnt pipeline).
// ---------------------------------------------------------------------------
__global__ __launch_bounds__(128) void attn_kernel(
    const f16* __restrict__ qb, const f16* __restrict__ kb,
    const f16* __restrict__ vbT, f16* __restrict__ ob)
{
    __shared__ __align__(16) f16 lq[64 * 64];          // [qrow][d]
    __shared__ __align__(16) f16 lk[2][64 * 64];       // [key][d]  (B-layout, QK^T)
    __shared__ __align__(16) f16 lv[2][64 * 64];       // [d][key]  (B-layout, PV)
    __shared__ __align__(16) f16 lp[4][16 * 64];       // wave-private P tiles

    const int tid = threadIdx.x, lane = tid & 31, w = tid >> 5;
    const int q0 = blockIdx.x * 64;
    const size_t head = ((size_t)blockIdx.z * NH + blockIdx.y) * (size_t)LSEQ * HD;
    const f16* Q = qb + head;
    const f16* K = kb + head;
    const f16* VT = vbT + head;                        // [d][l] per head

    // prefetch tile 0 (8 async ops per wave: 4 for K, 4 for V^T)
#pragma unroll
    for (int i = tid * 8; i < 64 * 64; i += 128 * 8)
        async_cp16(&lk[0][i], &K[i]);
#pragma unroll
    for (int i = tid * 8; i < 64 * 64; i += 128 * 8) {
        int d = i >> 6, key = i & 63;
        async_cp16(&lv[0][i], &VT[(size_t)d * LSEQ + key]);
    }
    for (int i = tid * 8; i < 64 * 64; i += 128 * 8)
        *(v8h*)&lq[i] = *(const v8h*)&Q[(size_t)q0 * HD + i];

    const f16* qrow = &lq[(w * 16 + (lane & 15)) * 64];
    v8f accO[4] = {};
    float mrow[8], lrow[8];
#pragma unroll
    for (int r = 0; r < 8; ++r) { mrow[r] = -1e30f; lrow[r] = 0.f; }
    const int hi8 = (lane >> 4) * 8;

    async_wait0();
    __syncthreads();
    v16h qa0 = load_a_frag(qrow, lane, 0);
    v16h qa1 = load_a_frag(qrow, lane, 32);

    int buf = 0;
    for (int kt = 0; kt < LSEQ; kt += 64, buf ^= 1) {
        if (kt + 64 < LSEQ) {   // prefetch next tile into other buffer
            const int nb = buf ^ 1, ktn = kt + 64;
#pragma unroll
            for (int i = tid * 8; i < 64 * 64; i += 128 * 8)
                async_cp16(&lk[nb][i], &K[(size_t)ktn * HD + i]);
#pragma unroll
            for (int i = tid * 8; i < 64 * 64; i += 128 * 8) {
                int d = i >> 6, key = i & 63;
                async_cp16(&lv[nb][i], &VT[(size_t)d * LSEQ + ktn + key]);
            }
            async_wait8();      // retire current tile's 8 ops, keep next in flight
        } else {
            async_wait0();
        }
        __syncthreads();

        // S = q @ k^T  (1/sqrt(D) already folded into q)
        v8f s[4] = {};
#pragma unroll
        for (int t = 0; t < 4; ++t) {
            s[t] = wmma_f16(qa0, load_b_frag(lk[buf], lane, 64, t * 16, 0),  s[t]);
            s[t] = wmma_f16(qa1, load_b_frag(lk[buf], lane, 64, t * 16, 32), s[t]);
        }
        // online softmax: each row spans a 16-lane half-wave
        float mnew[8], scale[8];
#pragma unroll
        for (int r = 0; r < 8; ++r) {
            float mv = fmaxf(fmaxf(s[0][r], s[1][r]), fmaxf(s[2][r], s[3][r]));
#pragma unroll
            for (int off = 1; off < 16; off <<= 1) mv = fmaxf(mv, __shfl_xor(mv, off, 32));
            mnew[r]  = fmaxf(mrow[r], mv);
            scale[r] = __expf(mrow[r] - mnew[r]);
        }
#pragma unroll
        for (int t = 0; t < 4; ++t)
#pragma unroll
            for (int r = 0; r < 8; ++r) {
                float p = __expf(s[t][r] - mnew[r]);
                s[t][r] = p;
                lp[w][(r + hi8) * 64 + t * 16 + (lane & 15)] = (f16)p;
            }
#pragma unroll
        for (int r = 0; r < 8; ++r) {
            float sum = s[0][r] + s[1][r] + s[2][r] + s[3][r];
#pragma unroll
            for (int off = 1; off < 16; off <<= 1) sum += __shfl_xor(sum, off, 32);
            lrow[r] = lrow[r] * scale[r] + sum;
            mrow[r] = mnew[r];
        }
#pragma unroll
        for (int t = 0; t < 4; ++t)
#pragma unroll
            for (int r = 0; r < 8; ++r) accO[t][r] *= scale[r];

        const f16* prow = &lp[w][(lane & 15) * 64];
        v16h pa0 = load_a_frag(prow, lane, 0);
        v16h pa1 = load_a_frag(prow, lane, 32);
#pragma unroll
        for (int t = 0; t < 4; ++t) {
            accO[t] = wmma_f16(pa0, load_b_frag(lv[buf], lane, 64, t * 16, 0),  accO[t]);
            accO[t] = wmma_f16(pa1, load_b_frag(lv[buf], lane, 64, t * 16, 32), accO[t]);
        }
        __syncthreads();        // all reads of buf done before it is refilled
    }

    f16* O = ob + head;
#pragma unroll
    for (int t = 0; t < 4; ++t)
#pragma unroll
        for (int r = 0; r < 8; ++r) {
            int row = q0 + w * 16 + r + hi8;
            int d   = t * 16 + (lane & 15);
            O[(size_t)row * HD + d] = (f16)(accO[t][r] / lrow[r]);
        }
}

// ---------------------------------------------------------------------------
// Kernel 3: out = O @ W_proj + b_proj  (O is f16 [B,H,L,D] -> logical [B*L, H*D])
// Block tile 128(M) x 64(N); f16 A-tile staged with async LDS copies.
// ---------------------------------------------------------------------------
__global__ __launch_bounds__(128) void proj_kernel(
    const f16* __restrict__ ob, const float* __restrict__ Wp,
    const float* __restrict__ bp, float* __restrict__ out)
{
    __shared__ __align__(16) f16 la[128 * 32];
    __shared__ __align__(16) f16 lb[64 * 32];
    const int tid = threadIdx.x, lane = tid & 31, w = tid >> 5;
    const int m0 = blockIdx.x * 128;
    const int n0 = blockIdx.y * 64;

    v8f acc[2][4] = {};
    for (int k0 = 0; k0 < DIM; k0 += 32) {
        if (k0 + 32 < DIM)
            __builtin_prefetch(Wp + (size_t)(k0 + 32) * DIM + n0, 0, 3);
        const int h = k0 >> 6, d0 = k0 & 63;           // head constant per 32-chunk
        for (int i = tid * 8; i < 128 * 32; i += 128 * 8) {
            int r = i >> 5, c = i & 31;
            int row = m0 + r, b = row >> 11, l = row & 2047;
            async_cp16(&la[r * 32 + c],
                       ob + ((((size_t)b * NH + h) * LSEQ + l) * HD + d0 + c));
        }
        for (int i = tid * 4; i < 32 * 64; i += 128 * 4) {
            int r = i >> 6, c = i & 63;
            float4 f = *(const float4*)(Wp + (size_t)(k0 + r) * DIM + n0 + c);
            lb[(c + 0) * 32 + r] = (f16)f.x; lb[(c + 1) * 32 + r] = (f16)f.y;
            lb[(c + 2) * 32 + r] = (f16)f.z; lb[(c + 3) * 32 + r] = (f16)f.w;
        }
        async_wait0();
        __syncthreads();
        v16h a0 = load_a_frag(&la[(w * 32 +      (lane & 15)) * 32], lane, 0);
        v16h a1 = load_a_frag(&la[(w * 32 + 16 + (lane & 15)) * 32], lane, 0);
#pragma unroll
        for (int t = 0; t < 4; ++t) {
            v16h b = load_b_frag(lb, lane, 32, t * 16, 0);
            acc[0][t] = wmma_f16(a0, b, acc[0][t]);
            acc[1][t] = wmma_f16(a1, b, acc[1][t]);
        }
        __syncthreads();
    }
    const int hi8 = (lane >> 4) * 8;
#pragma unroll
    for (int u = 0; u < 2; ++u)
#pragma unroll
        for (int t = 0; t < 4; ++t)
#pragma unroll
            for (int r = 0; r < 8; ++r) {
                int row = m0 + w * 32 + u * 16 + r + hi8;
                int n   = n0 + t * 16 + (lane & 15);
                out[(size_t)row * DIM + n] = acc[u][t][r] + bp[n];
            }
}

// ---------------------------------------------------------------------------
extern "C" void kernel_launch(void* const* d_in, const int* in_sizes, int n_in,
                              void* d_out, int out_size, void* d_ws, size_t ws_size,
                              hipStream_t stream) {
    (void)in_sizes; (void)n_in; (void)out_size; (void)ws_size;
    const float* x    = (const float*)d_in[0];
    const float* pe   = (const float*)d_in[1];
    const float* Wqkv = (const float*)d_in[2];
    const float* qs   = (const float*)d_in[3];
    const float* ks   = (const float*)d_in[4];
    const float* Wp   = (const float*)d_in[5];
    const float* bp   = (const float*)d_in[6];
    float* out = (float*)d_out;

    const size_t HSZ = (size_t)NB * NH * LSEQ * HD;    // 8,388,608 halves per tensor
    f16* ws   = (f16*)d_ws;
    f16* qbuf = ws;
    f16* kbuf = ws + HSZ;
    f16* vbuf = ws + 2 * HSZ;                          // transposed [B,H,D,L]
    f16* obuf = ws + 3 * HSZ;

    qkv_kernel<<<dim3(64, 48), 128, 0, stream>>>(x, Wqkv, pe, qs, ks, qbuf, kbuf, vbuf);
    attn_kernel<<<dim3(LSEQ / 64, NH, NB), 128, 0, stream>>>(qbuf, kbuf, vbuf, obuf);
    proj_kernel<<<dim3(64, 16), 128, 0, stream>>>(obuf, Wp, bp, out);
}
